// MultiHeadAttention_64647847739725
// MI455X (gfx1250) — matrix operand order repaired and checked by reading, not compiled, not run on hardware
//
#include <hip/hip_runtime.h>
#include <stdint.h>

// ---------------------------------------------------------------------------
// Types for CDNA5 WMMA (gfx1250, wave32)
// ---------------------------------------------------------------------------
typedef __bf16 bf16;
typedef bf16  v16bf __attribute__((ext_vector_type(16)));
typedef bf16  v8bf  __attribute__((ext_vector_type(8)));
typedef float v8f   __attribute__((ext_vector_type(8)));
typedef float v4f   __attribute__((ext_vector_type(4)));
typedef unsigned short v8us __attribute__((ext_vector_type(8)));
typedef unsigned short v4us __attribute__((ext_vector_type(4)));

// int4 in gcc-vector form, matching the async-LDS builtin prototype
typedef int v4i __attribute__((vector_size(16)));
typedef __attribute__((address_space(1))) v4i* gbl4_t;   // global (device) int4*
typedef __attribute__((address_space(3))) v4i* lds4_t;   // LDS (shared) int4*

#if __has_builtin(__builtin_amdgcn_global_load_async_to_lds_b128) && \
    __has_builtin(__builtin_amdgcn_s_wait_asynccnt)
#define HAVE_ASYNC_LDS 1
#else
#define HAVE_ASYNC_LDS 0
#endif

__device__ __forceinline__ unsigned short f2bf(float f) {
  unsigned u = __builtin_bit_cast(unsigned, f);
  u += 0x7FFFu + ((u >> 16) & 1u);          // round-to-nearest-even
  return (unsigned short)(u >> 16);
}

// ---------------------------------------------------------------------------
// fp32 -> bf16 conversion (vectorized x4)
// ---------------------------------------------------------------------------
__global__ void cvt_f32_to_bf16(const float* __restrict__ in,
                                unsigned short* __restrict__ out,
                                long long n4) {
  long long i = (long long)blockIdx.x * blockDim.x + threadIdx.x;
  if (i >= n4) return;
  v4f v = ((const v4f*)in)[i];
  v4us o;
  o[0] = f2bf(v[0]); o[1] = f2bf(v[1]); o[2] = f2bf(v[2]); o[3] = f2bf(v[3]);
  ((v4us*)out)[i] = o;
}

// ---------------------------------------------------------------------------
// Cooperative fill of one K-step B tile into LDS.
// Thread t copies 32B (one WMMA B fragment chunk) for (tile = t>>5, lane = t&31).
// Async path: GLOBAL_LOAD_ASYNC_TO_LDS_B128 x2 (ASYNCcnt DMA, overlaps WMMA).
// Fallback: vector global load + ds_store.
// ---------------------------------------------------------------------------
__device__ __forceinline__ void fill_B(unsigned short (*__restrict__ dst)[16],
                                       const bf16* __restrict__ src) {
  unsigned short* d = &dst[threadIdx.x][0];
#if HAVE_ASYNC_LDS
  gbl4_t g = (gbl4_t)src;     // C-style cast: const-strip + addrspacecast to AS1
  lds4_t l = (lds4_t)d;       // addrspacecast generic -> AS3
  // INST_OFFSET applies to both global and LDS addresses (ISA 08_async_tensor)
  __builtin_amdgcn_global_load_async_to_lds_b128(g, l, 0, 0);
  __builtin_amdgcn_global_load_async_to_lds_b128(g, l, 16, 0);
#else
  v16bf x = *(const v16bf*)src;
  *(v16bf*)d = x;
#endif
}

// ---------------------------------------------------------------------------
// Batched NT GEMM via v_wmma_f32_16x16x32_bf16:
//   C[z][m][n] = scale * ( sum_k A[z][m][k] * B[z][n][k]  + bias )
// A: bf16 row-major (lda), B: bf16 row-major over K (ldb), C: bf16 or f32 (ldc)
// BIAS: 0 = none, 1 = bias[n], 2 = bias[m]
// Block: 256 threads (8 waves), 128x128 tile; wave = 32x64 (2x4 WMMA tiles).
// B tile staged in LDS (double-buffered, 4x intra-WG reuse); A loaded direct.
// M, N multiples of 128; K multiple of 32 (all our shapes comply).
// ---------------------------------------------------------------------------
template<bool OUTBF, int BIAS>
__global__ __launch_bounds__(256) void gemm_nt_bf16(
    const unsigned short* __restrict__ Au, const unsigned short* __restrict__ Bu,
    void* __restrict__ Cv, const float* __restrict__ bias,
    int K, int lda, int ldb, int ldc,
    long long bsA, long long bsB, long long bsC, float scale)
{
  // [buf][chunk = tile*32 + lane][16 ushorts = 32B fragment]
  __shared__ __align__(32) unsigned short sB[2][256][16];

  const bf16* A  = (const bf16*)Au + (long long)blockIdx.z * bsA;
  const bf16* Bm = (const bf16*)Bu + (long long)blockIdx.z * bsB;

  const int tid  = threadIdx.x;
  const int w    = tid >> 5;
  const int lane = tid & 31;
  const int r    = lane & 15;       // row-in-tile (A) / col-in-tile (B,C)
  const int h    = lane >> 4;       // lane half
  const int wm   = w >> 1;          // 4 wave rows
  const int wn   = w & 1;           // 2 wave cols

  const long long m0    = (long long)blockIdx.x * 128 + wm * 32;
  const long long n0    = (long long)blockIdx.y * 128 + wn * 64;
  const long long n0blk = (long long)blockIdx.y * 128;

  const bf16* arow[2];
  arow[0] = A + (m0 +  0 + r) * (long long)lda;
  arow[1] = A + (m0 + 16 + r) * (long long)lda;

  // Per-thread B fill source: thread t -> (tile = t>>5, col c, half hh)
  const int fnt = tid >> 5, fc = tid & 15, fh = (tid >> 4) & 1;
  const bf16* bsrc = Bm + (n0blk + fnt * 16 + fc) * (long long)ldb + fh * 16;

  const v8f vzero = {0.f, 0.f, 0.f, 0.f, 0.f, 0.f, 0.f, 0.f};
  v8f acc[2][4];
#pragma unroll
  for (int mt = 0; mt < 2; ++mt)
#pragma unroll
    for (int nt = 0; nt < 4; ++nt) acc[mt][nt] = vzero;

  // Prologue: stage first B tile
  fill_B(sB[0], bsrc);
#if HAVE_ASYNC_LDS
  __builtin_amdgcn_s_wait_asynccnt(0);
#endif
  __syncthreads();

  int cur = 0;
  for (int kk = 0; kk < K; kk += 32) {
    // Kick off DMA of the next B tile while this K-step computes
    if (kk + 32 < K) fill_B(sB[cur ^ 1], bsrc + (kk + 32));

    // A fragment (direct global): lane half h holds K = kk+16h..+15
    union Frag { v16bf v; v8bf hlf[2]; } fa[2];
#pragma unroll
    for (int mt = 0; mt < 2; ++mt) {
      fa[mt].hlf[0] = *(const v8bf*)(arow[mt] + kk +     h * 16);
      fa[mt].hlf[1] = *(const v8bf*)(arow[mt] + kk + 8 + h * 16);
    }
    if (kk + 256 < K) {
#pragma unroll
      for (int mt = 0; mt < 2; ++mt) __builtin_prefetch(arow[mt] + kk + 256, 0, 1);
    }

    // B fragments from LDS (wave wn uses tiles wn*4 .. wn*4+3)
    v16bf fb[4];
#pragma unroll
    for (int nt = 0; nt < 4; ++nt)
      fb[nt] = *(const v16bf*)&sB[cur][(wn * 4 + nt) * 32 + lane][0];

#pragma unroll
    for (int mt = 0; mt < 2; ++mt)
#pragma unroll
      for (int nt = 0; nt < 4; ++nt)
        acc[mt][nt] = __builtin_amdgcn_wmma_f32_16x16x32_bf16(
            false, fa[mt].v, false, fb[nt], (short)0, acc[mt][nt], false, false);

#if HAVE_ASYNC_LDS
    __builtin_amdgcn_s_wait_asynccnt(0);   // next-tile DMA landed
#endif
    __syncthreads();
    cur ^= 1;
  }

  // Epilogue: C/D layout — vgpr j, lane r+16h -> row m0+16mt+j+8h, col n0+16nt+r
  unsigned short* Cb = (unsigned short*)Cv + (long long)blockIdx.z * bsC;
  float*          Cf = (float*)Cv          + (long long)blockIdx.z * bsC;

#pragma unroll
  for (int mt = 0; mt < 2; ++mt) {
#pragma unroll
    for (int nt = 0; nt < 4; ++nt) {
      const long long col = n0 + nt * 16 + r;
      float bn = 0.f;
      if (BIAS == 1) bn = bias[col];
      v8f c = acc[mt][nt];
#pragma unroll
      for (int j = 0; j < 8; ++j) {
        const long long row = m0 + mt * 16 + j + 8 * h;
        float v = c[j];
        if (BIAS == 1) v += bn;
        if (BIAS == 2) v += bias[row];
        v *= scale;
        const long long idx = row * (long long)ldc + col;
        if (OUTBF) Cb[idx] = f2bf(v);
        else       Cf[idx] = v;
      }
    }
  }
}

// ---------------------------------------------------------------------------
// Row softmax over 2048 contiguous fp32 elements -> bf16 out (same layout).
// One 256-thread block per row; 8 elements/thread; LDS tree reductions.
// ---------------------------------------------------------------------------
__global__ __launch_bounds__(256) void softmax_rows_2048(
    const float* __restrict__ S, unsigned short* __restrict__ out)
{
  __shared__ float red[256];
  const long long row = blockIdx.x;
  const float* p = S + row * 2048;
  const int t = threadIdx.x;

  v4f a = ((const v4f*)p)[2 * t];
  v4f b = ((const v4f*)p)[2 * t + 1];
  float x[8] = {a[0], a[1], a[2], a[3], b[0], b[1], b[2], b[3]};

  float m = x[0];
#pragma unroll
  for (int i = 1; i < 8; ++i) m = fmaxf(m, x[i]);
  red[t] = m; __syncthreads();
  for (int s2 = 128; s2 > 0; s2 >>= 1) {
    if (t < s2) red[t] = fmaxf(red[t], red[t + s2]);
    __syncthreads();
  }
  m = red[0]; __syncthreads();

  float e[8], sum = 0.f;
#pragma unroll
  for (int i = 0; i < 8; ++i) { e[i] = __expf(x[i] - m); sum += e[i]; }
  red[t] = sum; __syncthreads();
  for (int s2 = 128; s2 > 0; s2 >>= 1) {
    if (t < s2) red[t] += red[t + s2];
    __syncthreads();
  }
  const float inv = 1.0f / red[0];

  v8us o;
#pragma unroll
  for (int i = 0; i < 8; ++i) o[i] = f2bf(e[i] * inv);
  ((v8us*)(out + row * 2048))[t] = o;
}

// ---------------------------------------------------------------------------
// Host-side GEMM dispatcher
// ---------------------------------------------------------------------------
static void run_gemm(const unsigned short* A, const unsigned short* B, void* C,
                     const float* bias, int M, int N, int K,
                     int lda, int ldb, int ldc,
                     long long bsA, long long bsB, long long bsC, int batch,
                     float scale, bool outbf, int biasmode, hipStream_t st)
{
  dim3 grid(M / 128, N / 128, batch), blk(256);
#define GO(OB, BM) gemm_nt_bf16<OB, BM><<<grid, blk, 0, st>>>( \
      A, B, C, bias, K, lda, ldb, ldc, bsA, bsB, bsC, scale)
  if (outbf) {
    if      (biasmode == 0) GO(true, 0);
    else if (biasmode == 1) GO(true, 1);
    else                    GO(true, 2);
  } else {
    if      (biasmode == 0) GO(false, 0);
    else if (biasmode == 1) GO(false, 1);
    else                    GO(false, 2);
  }
#undef GO
}

// ---------------------------------------------------------------------------
// Launch: full attention pipeline
// ---------------------------------------------------------------------------
extern "C" void kernel_launch(void* const* d_in, const int* in_sizes, int n_in,
                              void* d_out, int out_size, void* d_ws, size_t ws_size,
                              hipStream_t stream) {
  (void)in_sizes; (void)n_in; (void)out_size; (void)ws_size;

  const float* Kin = (const float*)d_in[0];
  const float* Vin = (const float*)d_in[1];
  const float* Qin = (const float*)d_in[2];
  const float* Wk  = (const float*)d_in[3];  const float* bk = (const float*)d_in[4];
  const float* Wv  = (const float*)d_in[5];  const float* bv = (const float*)d_in[6];
  const float* Wq  = (const float*)d_in[7];  const float* bq = (const float*)d_in[8];
  const float* Wo  = (const float*)d_in[9];  const float* bo = (const float*)d_in[10];
  float* O = (float*)d_out;

  const long long Bz = 4, S = 2048, D = 1024, BS = Bz * S;   // 8192
  const long long MB = 1024 * 1024;
  char* ws = (char*)d_ws;

  unsigned short* Kb   = (unsigned short*)(ws +   0 * MB);   // 16 MB
  unsigned short* Vb   = (unsigned short*)(ws +  16 * MB);   // 16 MB
  unsigned short* Qb   = (unsigned short*)(ws +  32 * MB);   // 16 MB
  unsigned short* Wkb  = (unsigned short*)(ws +  48 * MB);   //  2 MB
  unsigned short* Wvb  = (unsigned short*)(ws +  50 * MB);
  unsigned short* Wqb  = (unsigned short*)(ws +  52 * MB);
  unsigned short* Wob  = (unsigned short*)(ws +  54 * MB);
  unsigned short* Kp   = (unsigned short*)(ws +  56 * MB);   // 16 MB [8192][1024]
  unsigned short* Qp   = (unsigned short*)(ws +  72 * MB);   // 16 MB [8192][1024] (pre-scaled)
  unsigned short* VpT  = (unsigned short*)(ws +  88 * MB);   // 16 MB [1024][8192] (V transposed)
  float*          Sc   = (float*)         (ws + 104 * MB);   // 64 MB [8192][2048] scores^T
  // reuse (dead after projections):
  unsigned short* alphaT = (unsigned short*)(ws +  0 * MB);  // 32 MB [8192][2048]
  unsigned short* Vacc   = (unsigned short*)(ws + 32 * MB);  // 16 MB [8192][1024]

  // 1) fp32 -> bf16 conversions
  const long long nIn4 = BS * D / 4;   // 2097152
  const long long nW4  = D * D / 4;    //  262144
  cvt_f32_to_bf16<<<dim3(nIn4 / 256), dim3(256), 0, stream>>>(Kin, Kb, nIn4);
  cvt_f32_to_bf16<<<dim3(nIn4 / 256), dim3(256), 0, stream>>>(Vin, Vb, nIn4);
  cvt_f32_to_bf16<<<dim3(nIn4 / 256), dim3(256), 0, stream>>>(Qin, Qb, nIn4);
  cvt_f32_to_bf16<<<dim3(nW4  / 256), dim3(256), 0, stream>>>(Wk, Wkb, nW4);
  cvt_f32_to_bf16<<<dim3(nW4  / 256), dim3(256), 0, stream>>>(Wv, Wvb, nW4);
  cvt_f32_to_bf16<<<dim3(nW4  / 256), dim3(256), 0, stream>>>(Wq, Wqb, nW4);
  cvt_f32_to_bf16<<<dim3(nW4  / 256), dim3(256), 0, stream>>>(Wo, Wob, nW4);

  // 2) Projections
  // Kp[s,e] = sum_d _K[s,d]*Wk[e,d] + bk[e]
  run_gemm(Kb, Wkb, Kp, bk, (int)BS, (int)D, (int)D, 1024, 1024, 1024,
           0, 0, 0, 1, 1.0f, /*bf16*/true, /*biasN*/1, stream);
  // Qp[s,e] = (sum_d _Q[s,d]*Wq[e,d] + bq[e]) / sqrt(D)
  run_gemm(Qb, Wqb, Qp, bq, (int)BS, (int)D, (int)D, 1024, 1024, 1024,
           0, 0, 0, 1, 0.03125f, true, 1, stream);
  // VpT[e,s] = sum_d Wv[e,d]*_V[s,d] + bv[e]   (V projected AND transposed)
  run_gemm(Wvb, Vb, VpT, bv, (int)D, (int)BS, (int)D, 1024, 1024, 8192,
           0, 0, 0, 1, 1.0f, true, /*biasM*/2, stream);

  // 3) Scores^T: Sc[b][q][k] = sum_e Qp[b,q,e]*Kp[b,k,e]   (fp32)
  run_gemm(Qp, Kp, Sc, nullptr, (int)S, (int)S, (int)D, 1024, 1024, 2048,
           S * D, S * D, S * S, (int)Bz, 1.0f, /*f32*/false, 0, stream);

  // 4) Softmax over k (contiguous rows) -> alphaT bf16
  softmax_rows_2048<<<dim3((unsigned)BS), dim3(256), 0, stream>>>(Sc, alphaT);

  // 5) V_[b][q][d] = sum_k alphaT[b,q,k] * VpT[d, b*2048+k]
  run_gemm(alphaT, VpT, Vacc, nullptr, (int)S, (int)D, (int)S, 2048, 8192, 1024,
           S * S, /*bsB=col offset*/S, S * D, (int)Bz, 1.0f, true, 0, stream);

  // 6) O[q,e] = sum_d V_[q,d]*Wo[e,d] + bo[e]   (fp32 to d_out)
  run_gemm(Vacc, Wob, O, bo, (int)BS, (int)D, (int)D, 1024, 1024, 1024,
           0, 0, 0, 1, 1.0f, /*f32*/false, /*biasN*/1, stream);
}
